// Block_25434796327403
// MI455X (gfx1250) — matrix operand order repaired and checked by reading, not compiled, not run on hardware
//
#include <hip/hip_runtime.h>
#include <hip/hip_bf16.h>

#define DIM    768
#define HEADS  12
#define HD     64
#define HIDDEN 3072
#define BB     8
#define NN     1024
#define MTOT   (BB*NN)        // 8192 rows
#define EPS    1e-6f

typedef __attribute__((ext_vector_type(16))) _Float16 v16h;
typedef __attribute__((ext_vector_type(8)))  _Float16 v8h;
typedef __attribute__((ext_vector_type(8)))  float    v8f;
typedef __attribute__((ext_vector_type(4)))  unsigned u32x4;
typedef __attribute__((ext_vector_type(8)))  unsigned u32x8;

static __device__ __forceinline__ v16h cat8(v8h lo, v8h hi) {
    return __builtin_shufflevector(lo, hi, 0,1,2,3,4,5,6,7,8,9,10,11,12,13,14,15);
}
static __device__ __forceinline__ v8h ld8(const _Float16* p) {
    return *(const v8h*)p;
}
static __device__ __forceinline__ v8f wmma_f16(v16h a, v16h b, v8f c) {
    return __builtin_amdgcn_wmma_f32_16x16x32_f16(false, a, false, b, (short)0, c,
                                                  false, false);
}
// lower 32 bits of a generic pointer into LDS == byte offset in the workgroup's LDS
static __device__ __forceinline__ unsigned lds_off(const void* p) {
    return (unsigned)(size_t)p;
}

// ---- Tensor Data Mover: 2-D tile (tile_rows x tile_k f16) from global to LDS ----
// D# per CDNA5 ISA ch.8: group0 (count=1, lds_addr, global_addr, type=2),
// group1 (data_size=2B, tensor dims, tile dims, row stride). VADDR2/3 = NULL (2D).
static __device__ __forceinline__ void tdm_load_2d(unsigned ldsoff,
                                                   const _Float16* gptr,
                                                   unsigned rows_total,
                                                   unsigned k_total,
                                                   unsigned tile_k,
                                                   unsigned tile_rows,
                                                   unsigned row_stride_elems) {
    unsigned long long ga = (unsigned long long)(size_t)gptr;
    u32x4 g0;
    g0[0] = 1u;                                            // count=1, user D#
    g0[1] = ldsoff;                                        // lds_addr (bytes)
    g0[2] = (unsigned)ga;                                  // global_addr[31:0]
    g0[3] = (unsigned)((ga >> 32) & 0x1FFFFFFu) | (2u << 30); // addr[56:32] | type=2
    u32x8 g1;
    g1[0] = (1u << 16);                                    // data_size=1 -> 2 bytes
    g1[1] = (k_total & 0xFFFFu) << 16;                     // tensor_dim0[15:0]
    g1[2] = ((k_total >> 16) & 0xFFFFu)                    // tensor_dim0[31:16]
          | ((rows_total & 0xFFFFu) << 16);                // tensor_dim1[15:0]
    g1[3] = ((rows_total >> 16) & 0xFFFFu)                 // tensor_dim1[31:16]
          | ((tile_k & 0xFFFFu) << 16);                    // tile_dim0
    g1[4] = (tile_rows & 0xFFFFu);                         // tile_dim1 (tile_dim2=0)
    g1[5] = row_stride_elems;                              // tensor_dim0_stride[31:0]
    g1[6] = 0u;                                            // stride[47:32], dim1_stride lo
    g1[7] = 0u;
    asm volatile("tensor_load_to_lds %0, %1" :: "s"(g0), "s"(g1) : "memory");
}

// ---------------- f32 -> f16 conversion ----------------
__global__ __launch_bounds__(256) void cvt_f32_f16(const float* __restrict__ in,
                                                   _Float16* __restrict__ out, int n) {
    int i = blockIdx.x * 256 + threadIdx.x;
    if (i < n) out[i] = (_Float16)in[i];
}

// ---------------- generic WMMA GEMM with TDM-staged weights -------------------
// C[m,n] = sum_k A[m,k] * W[n,k];  K must be a multiple of 128.
// EPI 0: store f32 | EPI 1: +bias +res, f32 | EPI 2: +bias, GELU, f16
template <int EPI>
__global__ __launch_bounds__(128)
void gemm_wmma(const _Float16* __restrict__ A, const _Float16* __restrict__ W,
               int M, int N, int K,
               const float* __restrict__ bias, const float* __restrict__ res,
               float* __restrict__ Cf, _Float16* __restrict__ Ch) {
    constexpr int KT = 128;                      // K-panel
    __shared__ __align__(16) _Float16 Wt[2][64 * KT];   // 2 x 16 KB double buffer

    const int lane = threadIdx.x & 31;
    const int wave = threadIdx.x >> 5;
    const int l15  = lane & 15;
    const int hi   = (lane < 16) ? 0 : 1;
    const int m0   = blockIdx.y * 64 + wave * 16;
    const int n0   = blockIdx.x * 64;
    const int npan = K / KT;

    if (wave == 0)
        tdm_load_2d(lds_off(&Wt[0][0]), W + (size_t)n0 * K,
                    (unsigned)N, (unsigned)K, KT, 64, (unsigned)K);

    const _Float16* arow = A + (size_t)(m0 + l15) * K + hi * 8;
    v8f acc[4] = {};
    for (int p = 0; p < npan; ++p) {
        if (wave == 0) __builtin_amdgcn_s_wait_tensorcnt(0);  // panel p landed
        __syncthreads();             // everyone done reading buf[(p+1)&1], p data vis.
        if (wave == 0 && (p + 1) < npan)
            tdm_load_2d(lds_off(&Wt[(p + 1) & 1][0]),
                        W + (size_t)n0 * K + (size_t)(p + 1) * KT,
                        (unsigned)N, (unsigned)K, KT, 64, (unsigned)K);

        const _Float16* wt = &Wt[p & 1][0];
        const _Float16* ap = arow + p * KT;
#pragma unroll
        for (int kk = 0; kk < KT; kk += 32) {
            v16h a = cat8(ld8(ap + kk), ld8(ap + kk + 16));
#pragma unroll
            for (int t = 0; t < 4; ++t) {
                const _Float16* brow = wt + (t * 16 + l15) * KT + kk + hi * 8;
                v16h b = cat8(ld8(brow), ld8(brow + 16));
                acc[t] = wmma_f16(a, b, acc[t]);
            }
        }
    }
#pragma unroll
    for (int t = 0; t < 4; ++t) {
        const int col = n0 + t * 16 + l15;
#pragma unroll
        for (int r = 0; r < 8; ++r) {
            const int row = m0 + r + hi * 8;
            float v = acc[t][r];
            if (EPI == 1) {
                v += bias[col] + res[(size_t)row * N + col];
                Cf[(size_t)row * N + col] = v;
            } else if (EPI == 2) {
                v += bias[col];
                float g = 0.5f * v * (1.0f + erff(v * 0.70710678118654752f));
                Ch[(size_t)row * N + col] = (_Float16)g;
            } else {
                Cf[(size_t)row * N + col] = v;
            }
        }
    }
}

// ---------------- split qkv -> Q(scaled,f16), K(f16), V^T(f16) ----------------
__global__ __launch_bounds__(256)
void split_qkv(const float* __restrict__ qkv, _Float16* __restrict__ Qh,
               _Float16* __restrict__ Kh, _Float16* __restrict__ VTh) {
    size_t idx = (size_t)blockIdx.x * 256 + threadIdx.x;   // over 8192*2304
    if (idx >= (size_t)MTOT * 3 * DIM) return;
    int m = (int)(idx / (3 * DIM));
    int j = (int)(idx % (3 * DIM));
    int t = j / DIM;
    int r = j % DIM;
    int h = r / HD;
    int d = r % HD;
    int b = m / NN;
    int n = m % NN;
    float v = qkv[idx];
    size_t bh = (size_t)(b * HEADS + h);
    if (t == 0)       Qh[(bh * NN + n) * HD + d] = (_Float16)(v * 0.125f); // 64^-0.5
    else if (t == 1)  Kh[(bh * NN + n) * HD + d] = (_Float16)v;
    else              VTh[(bh * HD + d) * NN + n] = (_Float16)v;
}

// ---------------- fused attention: one wave handles 16 q-rows of one head -------
__global__ __launch_bounds__(32)
void attn_kernel(const _Float16* __restrict__ Qh, const _Float16* __restrict__ Kh,
                 const _Float16* __restrict__ VTh, _Float16* __restrict__ attn_h) {
    __shared__ float    S[16 * NN];        // 64 KB scores
    __shared__ _Float16 P[16 * NN];        // 32 KB probs (f16, unnormalized)
    __shared__ float    rowsum[16];

    const int lane = threadIdx.x & 31;
    const int l15  = lane & 15;
    const int hi   = (lane < 16) ? 0 : 1;
    const int qt   = blockIdx.x;           // 0..63
    const int h    = blockIdx.y;
    const int b    = blockIdx.z;
    const size_t bh = (size_t)(b * HEADS + h);
    const _Float16* Qb = Qh + bh * NN * HD;
    const _Float16* Kb = Kh + bh * NN * HD;
    const _Float16* Vt = VTh + bh * HD * NN;

    const _Float16* qrow = Qb + (size_t)(qt * 16 + l15) * HD + hi * 8;
    v16h a0 = cat8(ld8(qrow + 0),  ld8(qrow + 16));
    v16h a1 = cat8(ld8(qrow + 32), ld8(qrow + 48));

    // S = Q @ K^T
    for (int t = 0; t < 64; ++t) {
        const _Float16* krow = Kb + (size_t)(t * 16 + l15) * HD + hi * 8;
        v16h b0 = cat8(ld8(krow),      ld8(krow + 16));
        v16h b1 = cat8(ld8(krow + 32), ld8(krow + 48));
        v8f c = {};
        c = wmma_f16(a0, b0, c);
        c = wmma_f16(a1, b1, c);
#pragma unroll
        for (int r = 0; r < 8; ++r)
            S[(r + hi * 8) * NN + t * 16 + l15] = c[r];
    }
    __syncthreads();

    // softmax: lanes (l, l+16) share row l, each scans half the row
    {
        const float* Sr = &S[l15 * NN];
        const int base = hi * 512;
        float mx = -1e30f;
        for (int j = 0; j < 512; ++j) mx = fmaxf(mx, Sr[base + j]);
        mx = fmaxf(mx, __shfl_xor(mx, 16));
        float sum = 0.f;
        for (int j = 0; j < 512; ++j) {
            float e = __expf(Sr[base + j] - mx);
            sum += e;
            P[l15 * NN + base + j] = (_Float16)e;
        }
        sum += __shfl_xor(sum, 16);
        if (lane < 16) rowsum[lane] = sum;
    }
    __syncthreads();

    // O = P @ V  (B rows from V^T)
    v8f o[4] = {};
    for (int k0 = 0; k0 < NN; k0 += 32) {
        const _Float16* pr = &P[l15 * NN + k0 + hi * 8];
        v16h a = cat8(*(const v8h*)pr, *(const v8h*)(pr + 16));
#pragma unroll
        for (int t = 0; t < 4; ++t) {
            const _Float16* vrow = Vt + (size_t)(t * 16 + l15) * NN + k0 + hi * 8;
            v16h bb = cat8(ld8(vrow), ld8(vrow + 16));
            o[t] = wmma_f16(a, bb, o[t]);
        }
    }
#pragma unroll
    for (int t = 0; t < 4; ++t) {
        const int d = t * 16 + l15;
#pragma unroll
        for (int r = 0; r < 8; ++r) {
            const int row = qt * 16 + r + hi * 8;
            float v = o[t][r] / rowsum[r + hi * 8];
            attn_h[((size_t)b * NN + row) * DIM + h * HD + d] = (_Float16)v;
        }
    }
}

// ---------------- layernorm over DIM, writes f32 (+optional f16 copy) ----------
__global__ __launch_bounds__(256)
void layernorm_kernel(const float* __restrict__ in, const float* __restrict__ gamma,
                      const float* __restrict__ beta, float* __restrict__ outf,
                      _Float16* __restrict__ outh) {
    __shared__ float red0[8], red1[8];
    const int row = blockIdx.x;
    const float* x = in + (size_t)row * DIM;
    float s = 0.f, s2 = 0.f;
    for (int j = threadIdx.x; j < DIM; j += 256) { float v = x[j]; s += v; s2 += v * v; }
    for (int o = 16; o > 0; o >>= 1) { s += __shfl_xor(s, o); s2 += __shfl_xor(s2, o); }
    const int w = threadIdx.x >> 5;
    if ((threadIdx.x & 31) == 0) { red0[w] = s; red1[w] = s2; }
    __syncthreads();
    s = 0.f; s2 = 0.f;
    for (int i = 0; i < 8; ++i) { s += red0[i]; s2 += red1[i]; }
    const float mu  = s * (1.0f / DIM);
    const float var = s2 * (1.0f / DIM) - mu * mu;
    const float inv = rsqrtf(var + EPS);
    for (int j = threadIdx.x; j < DIM; j += 256) {
        float v = (x[j] - mu) * inv * gamma[j] + beta[j];
        outf[(size_t)row * DIM + j] = v;
        if (outh) outh[(size_t)row * DIM + j] = (_Float16)v;
    }
}

// ---------------- host orchestration ----------------
extern "C" void kernel_launch(void* const* d_in, const int* in_sizes, int n_in,
                              void* d_out, int out_size, void* d_ws, size_t ws_size,
                              hipStream_t stream) {
    const float* x      = (const float*)d_in[0];
    const float* w_qkv  = (const float*)d_in[1];
    const float* w_proj = (const float*)d_in[2];
    const float* b_proj = (const float*)d_in[3];
    const float* w_fc1  = (const float*)d_in[4];
    const float* b_fc1  = (const float*)d_in[5];
    const float* w_fc2  = (const float*)d_in[6];
    const float* b_fc2  = (const float*)d_in[7];
    const float* gamma1 = (const float*)d_in[8];
    const float* beta1  = (const float*)d_in[9];
    const float* gamma2 = (const float*)d_in[10];
    const float* beta2  = (const float*)d_in[11];
    float* out = (float*)d_out;

    char* wp = (char*)d_ws;
    auto alloc = [&](size_t bytes) {
        void* p = wp;
        wp += (bytes + 255) & ~(size_t)255;
        return p;
    };
    _Float16* xh     = (_Float16*)alloc((size_t)MTOT * DIM * 2);
    _Float16* wqkvh  = (_Float16*)alloc((size_t)3 * DIM * DIM * 2);
    _Float16* wprojh = (_Float16*)alloc((size_t)DIM * DIM * 2);
    _Float16* wfc1h  = (_Float16*)alloc((size_t)HIDDEN * DIM * 2);
    _Float16* wfc2h  = (_Float16*)alloc((size_t)DIM * HIDDEN * 2);
    float*    qkvf   = (float*)   alloc((size_t)MTOT * 3 * DIM * 4);
    _Float16* Qh     = (_Float16*)alloc((size_t)BB * HEADS * NN * HD * 2);
    _Float16* Kh     = (_Float16*)alloc((size_t)BB * HEADS * NN * HD * 2);
    _Float16* VTh    = (_Float16*)alloc((size_t)BB * HEADS * HD * NN * 2);
    _Float16* attnh  = (_Float16*)alloc((size_t)MTOT * DIM * 2);
    float*    y1     = (float*)   alloc((size_t)MTOT * DIM * 4);
    float*    x1     = (float*)   alloc((size_t)MTOT * DIM * 4);
    _Float16* x1h    = (_Float16*)alloc((size_t)MTOT * DIM * 2);
    _Float16* hh     = (_Float16*)alloc((size_t)MTOT * HIDDEN * 2);
    float*    y2     = (float*)   alloc((size_t)MTOT * DIM * 4);

    auto cvt = [&](const float* src, _Float16* dst, size_t n) {
        cvt_f32_f16<<<(unsigned)((n + 255) / 256), 256, 0, stream>>>(src, dst, (int)n);
    };
    cvt(x, xh, (size_t)MTOT * DIM);
    cvt(w_qkv, wqkvh, (size_t)3 * DIM * DIM);
    cvt(w_proj, wprojh, (size_t)DIM * DIM);
    cvt(w_fc1, wfc1h, (size_t)HIDDEN * DIM);
    cvt(w_fc2, wfc2h, (size_t)DIM * HIDDEN);

    // qkv = x @ w_qkv^T  : [8192, 2304]
    gemm_wmma<0><<<dim3(3 * DIM / 64, MTOT / 64), 128, 0, stream>>>(
        xh, wqkvh, MTOT, 3 * DIM, DIM, nullptr, nullptr, qkvf, nullptr);

    split_qkv<<<(unsigned)(((size_t)MTOT * 3 * DIM + 255) / 256), 256, 0, stream>>>(
        qkvf, Qh, Kh, VTh);

    attn_kernel<<<dim3(NN / 16, HEADS, BB), 32, 0, stream>>>(Qh, Kh, VTh, attnh);

    // y1 = attn @ w_proj^T + b_proj + x
    gemm_wmma<1><<<dim3(DIM / 64, MTOT / 64), 128, 0, stream>>>(
        attnh, wprojh, MTOT, DIM, DIM, b_proj, x, y1, nullptr);

    layernorm_kernel<<<MTOT, 256, 0, stream>>>(y1, gamma1, beta1, x1, x1h);

    // h = gelu(x1 @ w_fc1^T + b_fc1)  -> f16
    gemm_wmma<2><<<dim3(HIDDEN / 64, MTOT / 64), 128, 0, stream>>>(
        x1h, wfc1h, MTOT, HIDDEN, DIM, b_fc1, nullptr, nullptr, hh);

    // y2 = h @ w_fc2^T + b_fc2 + x1
    gemm_wmma<1><<<dim3(DIM / 64, MTOT / 64), 128, 0, stream>>>(
        hh, wfc2h, MTOT, DIM, HIDDEN, b_fc2, x1, y2, nullptr);

    layernorm_kernel<<<MTOT, 256, 0, stream>>>(y2, gamma2, beta2, out, nullptr);
}
